// BiLSTMLayer_62491774156959
// MI455X (gfx1250) — matrix operand order repaired and checked by reading, not compiled, not run on hardware
//
#include <hip/hip_runtime.h>
#include <hip/hip_bf16.h>

// ---------------- problem constants ----------------
#define Bq 128
#define Tq 256
#define Hq 512
#define G4 2048          // 4*H
#define KK 1024          // H (input) + H (hidden)
#define NKT 32           // KK / 32 k-tiles
#define NJT 32           // Hq / 16 column tiles

typedef __attribute__((ext_vector_type(16))) __bf16 v16bf;
typedef __attribute__((ext_vector_type(8)))  float  v8f;

union FragB { v16bf v; uint4 q[2]; };
struct Stage { FragB a, bI, bF, bG, bO; };

__device__ __forceinline__ __bf16 f2bf(float f) {
    unsigned u = __builtin_bit_cast(unsigned, f);
    u += 0x7FFFu + ((u >> 16) & 1u);            // round-to-nearest-even
    unsigned short s = (unsigned short)(u >> 16);
    return __builtin_bit_cast(__bf16, s);
}
__device__ __forceinline__ float sigmoidf_(float x) {
    return 1.0f / (1.0f + __expf(-x));
}
__device__ __forceinline__ float tanhf_(float x) {
    x = fminf(fmaxf(x, -15.0f), 15.0f);
    float e = __expf(2.0f * x);
    return (e - 1.0f) / (e + 1.0f);
}

// ---------------- fp32 -> bf16 convert ----------------
__global__ void cvt_f32_bf16(const float* __restrict__ in, __bf16* __restrict__ out, int n) {
    int i = blockIdx.x * blockDim.x + threadIdx.x;
    if (i < n) out[i] = f2bf(in[i]);
}

// ---------------- pack [Wx;Wh] into WMMA B-fragment layout ----------------
// Wp[l][d][kt][nt][lane][e], e=0..15 ; K = kt*32 + (e&8)*2 + (lane>>4)*8 + (e&7) ; n = nt*16 + (lane&15)
__global__ void pack_w(const float* __restrict__ Wx, const float* __restrict__ Wh,
                       __bf16* __restrict__ Wp) {
    int tid  = blockIdx.x * blockDim.x + threadIdx.x;   // < 2*2*1024*2048 = 8388608
    int e    = tid & 15;
    int lane = (tid >> 4) & 31;
    int nt   = (tid >> 9) & 127;
    int kt   = (tid >> 16) & 31;
    int d    = (tid >> 21) & 1;
    int l    = (tid >> 22) & 1;
    int K = kt * 32 + (e & 8) * 2 + (lane >> 4) * 8 + (e & 7);
    int n = nt * 16 + (lane & 15);
    float v;
    if (K < Hq) v = Wx[((size_t)(l * 2 + d) * Hq + K) * G4 + n];
    else        v = Wh[((size_t)(l * 2 + d) * Hq + (K - Hq)) * G4 + n];
    Wp[tid] = f2bf(v);
}

// ---------------- zero h (both ping-pong bufs) and c state ----------------
__global__ void zero_state(__bf16* __restrict__ h, float* __restrict__ c) {
    int i = blockIdx.x * blockDim.x + threadIdx.x;      // 262144 threads
    if (i < 2 * 2 * Bq * Hq) h[i] = __builtin_bit_cast(__bf16, (unsigned short)0);
    if (i < 2 * Bq * Hq)     c[i] = 0.0f;
}

// Load one software-pipeline stage: A fragment (16x32 bf16) + 4 gate B fragments.
__device__ __forceinline__ void load_stage(Stage& st, int kt,
                                           const __bf16* __restrict__ xrow,
                                           const __bf16* __restrict__ hrow,
                                           const __bf16* __restrict__ wlanebase, // incl. dir + lane*16
                                           int jt, int hs)
{
    // A layout (ISA 7.12.2): lanes0-15 K=[0..7],[16..23]; lanes16-31 K=[8..15],[24..31]
    const __bf16* arow = (kt < 16) ? (xrow + kt * 32) : (hrow + (kt - 16) * 32);
    st.a.q[0] = *(const uint4*)(arow + hs * 8);
    st.a.q[1] = *(const uint4*)(arow + hs * 8 + 16);

    const __bf16* wk = wlanebase + (size_t)kt * (128 * 32 * 16);
    const __bf16* p0 = wk + (size_t)(0 * 32 + jt) * 512;
    const __bf16* p1 = wk + (size_t)(1 * 32 + jt) * 512;
    const __bf16* p2 = wk + (size_t)(2 * 32 + jt) * 512;
    const __bf16* p3 = wk + (size_t)(3 * 32 + jt) * 512;
    st.bI.q[0] = *(const uint4*)(p0); st.bI.q[1] = *(const uint4*)(p0 + 8);
    st.bF.q[0] = *(const uint4*)(p1); st.bF.q[1] = *(const uint4*)(p1 + 8);
    st.bG.q[0] = *(const uint4*)(p2); st.bG.q[1] = *(const uint4*)(p2 + 8);
    st.bO.q[0] = *(const uint4*)(p3); st.bO.q[1] = *(const uint4*)(p3 + 8);
}

// ---------------- one LSTM timestep, both directions ----------------
// grid = (32 jt, 2 mhalf, 2 dir), block = 128 (4 waves, wave w owns rows mb*64 + w*16 ..)
__global__ __launch_bounds__(128) void lstm_step(
    const __bf16* __restrict__ in_seq, unsigned long long in_dir_stride,
    const __bf16* __restrict__ Wp,      // this layer: [dir][kt][nt][lane][16] bf16
    const float*  __restrict__ bias,    // this layer: [dir][2048]
    const __bf16* __restrict__ h_read,  // [dir][128][512]
    __bf16*       __restrict__ h_write, // [dir][128][512]
    float*        __restrict__ c_state, // [dir][128][512]
    __bf16*       __restrict__ seq_out, // layer0: [dir][128][256][512], else unused
    float*        __restrict__ final_out, // layer1: [128][256][1024], else null
    int s)
{
    const int w    = threadIdx.x >> 5;  // 0..3
    const int lane = threadIdx.x & 31;
    const int hs   = lane >> 4;         // K-half select for A, row-half for D
    const int l16  = lane & 15;
    const int jt   = blockIdx.x;        // 0..31 : 16-wide column tile of hidden dim
    const int mb   = blockIdx.y;        // 0..1  : batch half
    const int d    = blockIdx.z;        // direction
    const int t    = (d == 0) ? s : (Tq - 1 - s);

    const int rowA = mb * 64 + w * 16 + l16;   // batch index for A rows
    const __bf16* xrow = in_seq + (size_t)d * in_dir_stride + ((size_t)rowA * Tq + t) * Hq;
    const __bf16* hrow = h_read + ((size_t)d * Bq + rowA) * Hq;
    const __bf16* wlanebase = Wp + (size_t)d * NKT * 128 * 32 * 16 + (size_t)lane * 16;

    v8f accI = (v8f)0.0f, accF = (v8f)0.0f, accG = (v8f)0.0f, accO = (v8f)0.0f;

    Stage st[2];
    load_stage(st[0], 0, xrow, hrow, wlanebase, jt, hs);

#pragma unroll 2
    for (int kt = 0; kt < NKT; ++kt) {
        const int cur = kt & 1;
        if (kt + 1 < NKT)                       // prefetch next stage before consuming current
            load_stage(st[cur ^ 1], kt + 1, xrow, hrow, wlanebase, jt, hs);
        const Stage& c = st[cur];
        accI = __builtin_amdgcn_wmma_f32_16x16x32_bf16(false, c.a.v, false, c.bI.v, (short)0, accI, false, false);
        accF = __builtin_amdgcn_wmma_f32_16x16x32_bf16(false, c.a.v, false, c.bF.v, (short)0, accF, false, false);
        accG = __builtin_amdgcn_wmma_f32_16x16x32_bf16(false, c.a.v, false, c.bG.v, (short)0, accG, false, false);
        accO = __builtin_amdgcn_wmma_f32_16x16x32_bf16(false, c.a.v, false, c.bO.v, (short)0, accO, false, false);
    }

    // epilogue: bias + LSTM nonlinearities for this wave's 16x16 h tile
    const float* bp = bias + (size_t)d * G4 + jt * 16 + l16;
    const float bI_ = bp[0], bF_ = bp[512], bG_ = bp[1024], bO_ = bp[1536];
    const int j = jt * 16 + l16;

#pragma unroll
    for (int r = 0; r < 8; ++r) {
        const int rowD = mb * 64 + w * 16 + hs * 8 + r;     // D layout: M = r + 8*(lane>=16)
        const size_t cidx = ((size_t)d * Bq + rowD) * Hq + j;
        float iv = sigmoidf_(accI[r] + bI_);
        float fv = sigmoidf_(accF[r] + bF_);
        float gv = tanhf_(accG[r] + bG_);
        float ov = sigmoidf_(accO[r] + bO_);
        float cn = fv * c_state[cidx] + iv * gv;
        c_state[cidx] = cn;
        float hn = ov * tanhf_(cn);
        h_write[cidx] = f2bf(hn);
        if (final_out) {
            final_out[((size_t)rowD * Tq + t) * (2 * Hq) + (size_t)d * Hq + j] = hn;
        } else {
            seq_out[(size_t)d * ((size_t)Bq * Tq * Hq) + ((size_t)rowD * Tq + t) * Hq + j] = f2bf(hn);
        }
    }
}

// ---------------- host ----------------
extern "C" void kernel_launch(void* const* d_in, const int* in_sizes, int n_in,
                              void* d_out, int out_size, void* d_ws, size_t ws_size,
                              hipStream_t stream) {
    (void)in_sizes; (void)n_in; (void)out_size; (void)ws_size;
    const float* x  = (const float*)d_in[0];   // [128,256,512]
    const float* Wx = (const float*)d_in[1];   // [2,2,512,2048]
    const float* Wh = (const float*)d_in[2];   // [2,2,512,2048]
    const float* bb = (const float*)d_in[3];   // [2,2,2048]
    float* out = (float*)d_out;                // [128,256,1024]

    char* ws = (char*)d_ws;
    __bf16* xbf  = (__bf16*)(ws);                                      // 33554432 B
    __bf16* Wp   = (__bf16*)(ws + 33554432ULL);                        // 16777216 B
    __bf16* seq0 = (__bf16*)(ws + 33554432ULL + 16777216ULL);          // 67108864 B
    __bf16* hst  = (__bf16*)(ws + 33554432ULL + 16777216ULL + 67108864ULL);          // 524288 B
    float*  cst  = (float*) (ws + 33554432ULL + 16777216ULL + 67108864ULL + 524288ULL); // 524288 B

    cvt_f32_bf16<<<65536, 256, 0, stream>>>(x, xbf, Bq * Tq * Hq);
    pack_w<<<32768, 256, 0, stream>>>(Wx, Wh, Wp);

    const size_t HPP = (size_t)2 * Bq * Hq;  // ping-pong stride (elements)
    for (int l = 0; l < 2; ++l) {
        zero_state<<<1024, 256, 0, stream>>>(hst, cst);
        const __bf16* in_seq = (l == 0) ? xbf : seq0;
        unsigned long long stride = (l == 0) ? 0ULL : (unsigned long long)Bq * Tq * Hq;
        const __bf16* wl = Wp + (size_t)l * 2 * KK * G4;
        const float*  bl = bb + (size_t)l * 2 * G4;
        __bf16* so = (l == 0) ? seq0 : nullptr;
        float*  fo = (l == 0) ? nullptr : out;
        for (int s = 0; s < Tq; ++s) {
            const __bf16* hr = hst + (size_t)(s & 1) * HPP;
            __bf16*       hw = hst + (size_t)((s + 1) & 1) * HPP;
            lstm_step<<<dim3(NJT, 2, 2), 128, 0, stream>>>(in_seq, stride, wl, bl,
                                                           hr, hw, cst, so, fo, s);
        }
    }
}
